// Model_Mine_61624190763253
// MI455X (gfx1250) — matrix operand order repaired and checked by reading, not compiled, not run on hardware
//
#include <hip/hip_runtime.h>

typedef __attribute__((ext_vector_type(16))) _Float16 v16h;
typedef __attribute__((ext_vector_type(8)))  _Float16 v8h;
typedef __attribute__((ext_vector_type(8)))  float    v8f;

#define S_LEN 256
#define D_DIM 64
#define K_INT 4
#define C_NUM 512
#define B_SZ  512
#define NMID  500000
#define NEGF  (-4294967295.0f)
#define LDH   72    // f16 row stride: 144B = 16B aligned, 36n mod 64 injective
#define VTLD  264   // transposed-items stride: 528B, 16B aligned, odd dword mix
#define PLD   32    // P staging stride

// workspace layout (float indices)
#define WS_WM     0
#define WS_CAND0  256
#define WS_CAND   (WS_CAND0 + C_NUM*D_DIM)
#define WS_T0     (WS_CAND  + C_NUM*D_DIM)
#define WS_TRT    (WS_T0    + C_NUM*D_DIM)
#define WS_POSSUM (WS_TRT   + C_NUM*C_NUM)
#define WS_DWT    (WS_POSSUM + 64)          // dense_w^T as f16: 4096 halves

// ---------------- helpers -----------------------------------------------------

__device__ __forceinline__ v8f wmma_f16(v16h a, v16h b, v8f c) {
  return __builtin_amdgcn_wmma_f32_16x16x32_f16(false, a, false, b,
                                                (short)0, c, false, false);
}

__device__ __forceinline__ v16h cat8(v8h lo, v8h hi) {
  return __builtin_shufflevector(lo, hi, 0, 1, 2, 3, 4, 5, 6, 7,
                                 8, 9, 10, 11, 12, 13, 14, 15);
}

// fast reciprocal: single v_rcp_f32 (no precise-div expansion)
__device__ __forceinline__ float rcp_fast(float x) {
  return __builtin_amdgcn_rcpf(x);
}

// A-frag 16x32 f16 from row-major tile: contiguous 8+8 halves per lane.
__device__ __forceinline__ v16h load_a16(const _Float16* m, int ld, int r0,
                                         int k0, int lane) {
  const _Float16* p = m + (r0 + (lane & 15)) * ld + k0 + ((lane & 16) ? 8 : 0);
  v8h lo = *(const v8h*)p;
  v8h hi = *(const v8h*)(p + 16);
  return cat8(lo, hi);
}

// B-frag 32x16 (KxN) where source is laid out so K is the contiguous axis:
// B[k][n] = m[(n0 + n)*ld + k0 + k]   -> 8+8 halves per lane (table k order).
__device__ __forceinline__ v16h load_bK(const _Float16* m, int ld, int n0,
                                        int k0, int lane) {
  const _Float16* p = m + (n0 + (lane & 15)) * ld + k0 + ((lane & 16) ? 8 : 0);
  v8h lo = *(const v8h*)p;
  v8h hi = *(const v8h*)(p + 16);
  return cat8(lo, hi);
}

// branch-free tanh: 1 - 2*rcp(e^2x + 1)  (one v_exp + one v_rcp + fma)
__device__ __forceinline__ float tanh_fast(float x) {
  float cx = fminf(fmaxf(x, -15.f), 15.f);
  float e = __expf(2.f * cx);
  return 1.f - 2.f * rcp_fast(e + 1.f);
}

// ---------------- prep: zero wmean, dense_w^T -> f16, pos column sums --------

__global__ void k_prep(const float* __restrict__ dense_w,
                       const float* __restrict__ pos_emb,
                       float* __restrict__ ws) {
  int tid = threadIdx.x;
  if (tid < 256) ws[WS_WM + tid] = 0.f;
  _Float16* dwt = (_Float16*)(ws + WS_DWT);
  for (int i = tid; i < D_DIM * D_DIM; i += blockDim.x) {
    int k = i >> 6, n = i & 63;               // dense_w[k][n]
    dwt[n * 64 + k] = (_Float16)dense_w[i];   // store transposed
  }
  if (tid < 64) {
    float s = 0.f;
    for (int r = 0; r < S_LEN; ++r) s += pos_emb[r * 64 + tid];
    ws[WS_POSSUM + tid] = s;
  }
}

// ---------------- fused E sweep: att_w + wmean accumulation (HBM bound) ------

__global__ void k_wmean(const float* __restrict__ E,
                        const float* __restrict__ W2,
                        float* __restrict__ ws, int n) {
  __shared__ float sW2[4 * 64];
  __shared__ float sAcc[4 * 64];
  int tid = threadIdx.x;
  for (int i = tid; i < 256; i += blockDim.x) { sW2[i] = W2[i]; sAcc[i] = 0.f; }
  __syncthreads();

  int lane  = tid & 31;
  int wv    = tid >> 5;
  int gwave = blockIdx.x * (blockDim.x >> 5) + wv;
  int nwave = gridDim.x * (blockDim.x >> 5);
  int c0    = lane * 2;

  float acc[4][2] = {};
  for (int row = gwave; row < n; row += nwave) {
    float2 e = *(const float2*)(E + (size_t)row * 64 + c0);
    float a[4];
#pragma unroll
    for (int k = 0; k < 4; ++k) {
      float p = e.x * sW2[k * 64 + c0] + e.y * sW2[k * 64 + c0 + 1];
#pragma unroll
      for (int sh = 16; sh >= 1; sh >>= 1) p += __shfl_xor(p, sh, 32);
      a[k] = p;
    }
#pragma unroll
    for (int k = 0; k < 4; ++k) {
      acc[k][0] += a[k] * e.x;
      acc[k][1] += a[k] * e.y;
    }
  }
#pragma unroll
  for (int k = 0; k < 4; ++k) {
    atomicAdd(&sAcc[k * 64 + c0], acc[k][0]);
    atomicAdd(&sAcc[k * 64 + c0 + 1], acc[k][1]);
  }
  __syncthreads();
  for (int i = tid; i < 256; i += blockDim.x) atomicAdd(&ws[WS_WM + i], sAcc[i]);
}

// ---------------- small candidate math ---------------------------------------

__global__ void k_cand0(const float* __restrict__ W3, float* __restrict__ ws) {
  int i = blockIdx.x * blockDim.x + threadIdx.x;   // c*64 + d
  int c = i >> 6, d = i & 63;
  float s = 0.f;
#pragma unroll
  for (int k = 0; k < 4; ++k) s += W3[k * C_NUM + c] * ws[WS_WM + k * 64 + d];
  ws[WS_CAND0 + i] = s;
}

__global__ void k_cand_t0(const float* __restrict__ W1,
                          const float* __restrict__ W4,
                          float* __restrict__ ws) {
  int i = blockIdx.x * blockDim.x + threadIdx.x;
  int c = i >> 6, d = i & 63;
  const float* row = ws + WS_CAND0 + c * 64;
  float s1 = 0.f, s4 = 0.f;
#pragma unroll 8
  for (int e = 0; e < 64; ++e) {
    float r = row[e];
    s1 += r * W1[e * 64 + d];
    s4 += r * W4[e * 64 + d];
  }
  ws[WS_CAND + i] = tanh_fast(s1);
  ws[WS_T0 + i]   = s4;
}

// trans stored transposed: T[c'][c] = softmax_over_c( t0[c] . cand0[c'] )
__global__ void k_trans(float* __restrict__ ws) {
  __shared__ float srow[C_NUM];
  __shared__ float sc0[64];
  __shared__ float sred[256];
  int cp = blockIdx.x, tid = threadIdx.x;
  if (tid < 64) sc0[tid] = ws[WS_CAND0 + cp * 64 + tid];
  __syncthreads();
  for (int c = tid; c < C_NUM; c += 256) {
    const float* t0 = ws + WS_T0 + c * 64;
    float s = 0.f;
#pragma unroll 8
    for (int e = 0; e < 64; ++e) s += t0[e] * sc0[e];
    srow[c] = s;
  }
  __syncthreads();
  sred[tid] = fmaxf(srow[tid], srow[tid + 256]);
  __syncthreads();
  for (int off = 128; off > 0; off >>= 1) {
    if (tid < off) sred[tid] = fmaxf(sred[tid], sred[tid + off]);
    __syncthreads();
  }
  float m = sred[0];
  __syncthreads();
  float e0 = __expf(srow[tid] - m), e1 = __expf(srow[tid + 256] - m);
  srow[tid] = e0; srow[tid + 256] = e1;
  sred[tid] = e0 + e1;
  __syncthreads();
  for (int off = 128; off > 0; off >>= 1) {
    if (tid < off) sred[tid] += sred[tid + off];
    __syncthreads();
  }
  float inv = rcp_fast(sred[0]);
  ws[WS_TRT + cp * C_NUM + tid]       = srow[tid] * inv;
  ws[WS_TRT + cp * C_NUM + tid + 256] = srow[tid + 256] * inv;
}

// ---------------- per-batch sequence pipeline (flash attention, WMMA) --------

__global__ void __launch_bounds__(256) k_seq(
    const float* __restrict__ E, const float* __restrict__ W1,
    const float* __restrict__ pos, const float* __restrict__ dense_b,
    const int* __restrict__ mid_his, const int* __restrict__ mask,
    const int* __restrict__ mid_batch, const float* __restrict__ ws,
    float* __restrict__ out) {
  __shared__ _Float16 sm_x[S_LEN * LDH];    // x -> q (in place)
  __shared__ _Float16 sm_vt[D_DIM * VTLD];  // items^T (f16): vt[d][s]
  __shared__ _Float16 sm_seq[S_LEN * LDH];  // seq (f16)
  __shared__ _Float16 sm_pst[8 * 16 * PLD]; // per-wave P staging
  __shared__ float sm_maskf[S_LEN];
  __shared__ float sm_logits[C_NUM];
  __shared__ float sm_logits2[C_NUM];
  __shared__ float sred[256];
  __shared__ int   sri[256];
  __shared__ float sm_mean[64];
  __shared__ float sm_meanh[64];
  __shared__ float sm_aq[4 * 64];
  __shared__ float sm_w[4 * S_LEN];
  __shared__ float sm_wsum[4];
  __shared__ float sm_ueb[4 * 64];
  __shared__ float sm_item[64];
  __shared__ int   sm_idx[4];
  __shared__ float sm_den;
  __shared__ int   sm_ridx;

  const int tid  = threadIdx.x;
  const int lane = tid & 31;
  const int wv   = tid >> 5;            // 8 waves
  const int b    = blockIdx.x;
  const int nl   = lane & 15;
  const int hb   = (lane >> 4) & 1;     // row-half of the wave
  const _Float16* dwt = (const _Float16*)(ws + WS_DWT);

  // ---- phase A: mask + gather items (transposed f16), x = items + pos ----
  if (tid == 0) sm_den = 0.f;
  sm_maskf[tid] = (float)mask[b * S_LEN + tid];
  __syncthreads();
  atomicAdd(&sm_den, sm_maskf[tid]);
  {
    int mid = mid_his[b * S_LEN + tid];
    const float* er = E + (size_t)mid * 64;
    const float* pr = pos + tid * 64;
#pragma unroll 4
    for (int d = 0; d < 64; d += 4) {
      float4 e4 = *(const float4*)(er + d);
      float4 p4 = *(const float4*)(pr + d);
      sm_vt[(d + 0) * VTLD + tid] = (_Float16)e4.x;
      sm_vt[(d + 1) * VTLD + tid] = (_Float16)e4.y;
      sm_vt[(d + 2) * VTLD + tid] = (_Float16)e4.z;
      sm_vt[(d + 3) * VTLD + tid] = (_Float16)e4.w;
      sm_x[tid * LDH + d + 0] = (_Float16)(e4.x + p4.x);
      sm_x[tid * LDH + d + 1] = (_Float16)(e4.y + p4.y);
      sm_x[tid * LDH + d + 2] = (_Float16)(e4.z + p4.z);
      sm_x[tid * LDH + d + 3] = (_Float16)(e4.w + p4.w);
    }
  }
  __syncthreads();

  // ---- phase B: q = tanh(x @ dense_w + b), WMMA, in place per 16-row tile ----
#pragma unroll
  for (int t = 0; t < 2; ++t) {
    int r0 = (wv + t * 8) * 16;
    v16h Ax0 = load_a16(sm_x, LDH, r0, 0, lane);
    v16h Ax1 = load_a16(sm_x, LDH, r0, 32, lane);
#pragma unroll
    for (int nc = 0; nc < 4; ++nc) {
      v8f C;
      float bb = dense_b[nc * 16 + nl];
#pragma unroll
      for (int e = 0; e < 8; ++e) C[e] = bb;
      v16h Bw0 = load_bK(dwt, 64, nc * 16, 0, lane);
      v16h Bw1 = load_bK(dwt, 64, nc * 16, 32, lane);
      C = wmma_f16(Ax0, Bw0, C);
      C = wmma_f16(Ax1, Bw1, C);
#pragma unroll
      for (int e = 0; e < 8; ++e)
        sm_x[(r0 + e + hb * 8) * LDH + nc * 16 + nl] = (_Float16)tanh_fast(C[e]);
    }
  }
  __syncthreads();

  // ---- phase C: flash attention  seq = softmax(mask(q q^T / 8)) @ items ----
  {
    _Float16* P = sm_pst + wv * (16 * PLD);
#pragma unroll
    for (int t = 0; t < 2; ++t) {
      int q0 = (wv + t * 8) * 16;
      v16h Aq0 = load_a16(sm_x, LDH, q0, 0, lane);
      v16h Aq1 = load_a16(sm_x, LDH, q0, 32, lane);
      float mrow[8], lrow[8], mq[8];
      v8f O[4];
#pragma unroll
      for (int e = 0; e < 8; ++e) {
        mrow[e] = -1e30f;
        lrow[e] = 0.f;
        mq[e]   = sm_maskf[q0 + e + hb * 8];   // hoisted query-row mask
      }
#pragma unroll
      for (int dc = 0; dc < 4; ++dc)
#pragma unroll
        for (int e = 0; e < 8; ++e) O[dc][e] = 0.f;

      for (int kp = 0; kp < 8; ++kp) {
        int s0 = kp * 32;
        // B = q^T slices: B[k][n] = q[s0+n][d0+k]
        v16h B00 = load_bK(sm_x, LDH, s0, 0, lane);
        v16h B01 = load_bK(sm_x, LDH, s0, 32, lane);
        v16h B10 = load_bK(sm_x, LDH, s0 + 16, 0, lane);
        v16h B11 = load_bK(sm_x, LDH, s0 + 16, 32, lane);
        v8f S0 = {}, S1 = {};
        S0 = wmma_f16(Aq0, B00, S0);
        S0 = wmma_f16(Aq1, B01, S0);
        S1 = wmma_f16(Aq0, B10, S1);
        S1 = wmma_f16(Aq1, B11, S1);
        float mk0 = sm_maskf[s0 + nl];         // hoisted key-col masks
        float mk1 = sm_maskf[s0 + 16 + nl];
#pragma unroll
        for (int e = 0; e < 8; ++e) {
          float a0 = S0[e] * 0.125f, a1 = S1[e] * 0.125f;
          a0 = (mq[e] * mk0 == 0.f) ? NEGF : a0;
          a1 = (mq[e] * mk1 == 0.f) ? NEGF : a1;
          float tm = fmaxf(a0, a1);
#pragma unroll
          for (int sh = 1; sh <= 8; sh <<= 1)
            tm = fmaxf(tm, __shfl_xor(tm, sh, 32));
          float mn = fmaxf(mrow[e], tm);
          float al = __expf(mrow[e] - mn);
          mrow[e] = mn;
          float p0 = __expf(a0 - mn), p1 = __expf(a1 - mn);
          float ps = p0 + p1;
#pragma unroll
          for (int sh = 1; sh <= 8; sh <<= 1) ps += __shfl_xor(ps, sh, 32);
          lrow[e] = lrow[e] * al + ps;
#pragma unroll
          for (int dc = 0; dc < 4; ++dc) O[dc][e] *= al;
          P[(e + hb * 8) * PLD + nl]      = (_Float16)p0;
          P[(e + hb * 8) * PLD + 16 + nl] = (_Float16)p1;
        }
        // wave-private LDS staging: DS ops are in-order within a wave
        v16h Ap = load_a16(P, PLD, 0, 0, lane);
#pragma unroll
        for (int dc = 0; dc < 4; ++dc) {
          // B[k][n] = items[s0+k][dc*16+n] = vt[(dc*16+n)*VTLD + s0+k]
          v16h Bv = load_bK(sm_vt, VTLD, dc * 16, s0, lane);
          O[dc] = wmma_f16(Ap, Bv, O[dc]);
        }
      }
#pragma unroll
      for (int e = 0; e < 8; ++e) {
        int r = q0 + e + hb * 8;
        float scl = (lrow[e] > 0.f) ? (mq[e] * rcp_fast(lrow[e])) : 0.f;
#pragma unroll
        for (int dc = 0; dc < 4; ++dc)
          sm_seq[r * LDH + dc * 16 + nl] = (_Float16)(O[dc][e] * scl);
      }
    }
  }
  __syncthreads();

  // ---- phase D: pooling, logits, trans, top-4, interests, readout ----
  if (tid < 64) sm_mean[tid] = 0.f;
  __syncthreads();
  {
    int d = tid & 63, r0 = tid >> 6;
    float acc = 0.f;
    for (int r = r0; r < S_LEN; r += 4) acc += (float)sm_seq[r * LDH + d];
    atomicAdd(&sm_mean[d], acc);
  }
  __syncthreads();
  if (tid < 64) {
    float rdenom = rcp_fast(sm_den + 1e-9f);
    sm_mean[tid] = (sm_mean[tid] + ws[WS_POSSUM + tid]) * rdenom;
  }
  __syncthreads();
  if (tid < 64) {
    float s = 0.f;
#pragma unroll 8
    for (int e = 0; e < 64; ++e) s += sm_mean[e] * W1[e * 64 + tid];
    sm_meanh[tid] = tanh_fast(s);
  }
  __syncthreads();

  for (int c = tid; c < C_NUM; c += 256) {
    const float* cr = ws + WS_CAND + c * 64;
    float s = 0.f;
#pragma unroll 8
    for (int e = 0; e < 64; ++e) s += sm_meanh[e] * cr[e];
    sm_logits[c] = s;
  }
  __syncthreads();
  for (int c = tid; c < C_NUM; c += 256) {
    const float* tr = ws + WS_TRT + c * C_NUM;
    float s = 0.f;
#pragma unroll 8
    for (int e = 0; e < C_NUM; ++e) s += sm_logits[e] * tr[e];
    sm_logits2[c] = s;
  }
  __syncthreads();

  // top-4 (sequential argmax)
  for (int it = 0; it < 4; ++it) {
    float best = -1e30f; int bi = 0;
    for (int c = tid; c < C_NUM; c += 256) {
      float v = sm_logits2[c];
      if (v > best) { best = v; bi = c; }
    }
    sred[tid] = best; sri[tid] = bi;
    __syncthreads();
    for (int off = 128; off > 0; off >>= 1) {
      if (tid < off && sred[tid + off] > sred[tid]) {
        sred[tid] = sred[tid + off]; sri[tid] = sri[tid + off];
      }
      __syncthreads();
    }
    if (tid == 0) { sm_idx[it] = sri[0]; sm_logits2[sri[0]] = -1e30f; }
    __syncthreads();
  }
  {
    int k = tid >> 6, d = tid & 63;
    sm_aq[k * 64 + d] = ws[WS_CAND + sm_idx[k] * 64 + d];
  }
  __syncthreads();

  // masked interest attention logits: w[k][s] = tanh((seq+pos)@W1)[s] . aq[k]
  {
    int s = tid;
    float xr[64];
    const float* pr = pos + s * 64;
#pragma unroll
    for (int d = 0; d < 64; ++d) xr[d] = (float)sm_seq[s * LDH + d] + pr[d];
    float wk[4] = {0.f, 0.f, 0.f, 0.f};
    for (int d = 0; d < 64; ++d) {
      float ih = 0.f;
#pragma unroll 8
      for (int e = 0; e < 64; ++e) ih += xr[e] * W1[e * 64 + d];
      ih = tanh_fast(ih);
#pragma unroll
      for (int k = 0; k < 4; ++k) wk[k] += ih * sm_aq[k * 64 + d];
    }
    float mv = sm_maskf[s];
#pragma unroll
    for (int k = 0; k < 4; ++k)
      sm_w[k * S_LEN + s] = (mv == 0.f) ? NEGF : wk[k];
  }
  __syncthreads();
  for (int k = 0; k < 4; ++k) {
    sred[tid] = sm_w[k * S_LEN + tid];
    __syncthreads();
    for (int off = 128; off > 0; off >>= 1) {
      if (tid < off) sred[tid] = fmaxf(sred[tid], sred[tid + off]);
      __syncthreads();
    }
    float m = sred[0];
    __syncthreads();
    float p = __expf(sm_w[k * S_LEN + tid] - m);
    sm_w[k * S_LEN + tid] = p;
    sred[tid] = p;
    __syncthreads();
    for (int off = 128; off > 0; off >>= 1) {
      if (tid < off) sred[tid] += sred[tid + off];
      __syncthreads();
    }
    if (tid == 0) sm_wsum[k] = sred[0];
    __syncthreads();
  }
  {
    int k = tid >> 6, d = tid & 63;
    float s = 0.f;
    for (int r = 0; r < S_LEN; ++r)
      s += sm_w[k * S_LEN + r] * (float)sm_seq[r * LDH + d];
    float u = s * rcp_fast(sm_wsum[k]);
    sm_ueb[k * 64 + d] = u;
    out[B_SZ * 64 + (b * 4 + k) * 64 + d] = u;  // user_eb
  }
  if (tid < 64) sm_item[tid] = E[(size_t)mid_batch[b] * 64 + tid];
  __syncthreads();
  if (tid < 4) {
    float s = 0.f;
#pragma unroll 8
    for (int d = 0; d < 64; ++d) s += sm_ueb[tid * 64 + d] * sm_item[d];
    sred[tid] = s;   // softmax is monotone: argmax on raw dots
  }
  __syncthreads();
  if (tid == 0) {
    int ri = 0; float bv = sred[0];
#pragma unroll
    for (int k = 1; k < 4; ++k) if (sred[k] > bv) { bv = sred[k]; ri = k; }
    sm_ridx = ri;
  }
  __syncthreads();
  if (tid < 64) out[b * 64 + tid] = sm_ueb[sm_ridx * 64 + tid];       // readout
  if (tid < 4)
    out[B_SZ * 64 + B_SZ * 4 * 64 + b * 4 + tid] = (float)sm_idx[tid]; // idx
}

// ---------------- launch ------------------------------------------------------

extern "C" void kernel_launch(void* const* d_in, const int* in_sizes, int n_in,
                              void* d_out, int out_size, void* d_ws,
                              size_t ws_size, hipStream_t stream) {
  (void)in_sizes; (void)n_in; (void)out_size; (void)ws_size;
  const float* E   = (const float*)d_in[0];
  const float* W1  = (const float*)d_in[1];
  const float* W2  = (const float*)d_in[2];
  const float* W3  = (const float*)d_in[3];
  const float* W4  = (const float*)d_in[4];
  const float* pos = (const float*)d_in[5];
  const float* dw  = (const float*)d_in[6];
  const float* db  = (const float*)d_in[7];
  const int*   mh  = (const int*)d_in[8];
  const int*   mk  = (const int*)d_in[9];
  const int*   mb  = (const int*)d_in[10];
  float* ws  = (float*)d_ws;
  float* out = (float*)d_out;

  k_prep<<<1, 256, 0, stream>>>(dw, pos, ws);
  k_wmean<<<1024, 256, 0, stream>>>(E, W2, ws, NMID);
  k_cand0<<<(C_NUM * D_DIM) / 256, 256, 0, stream>>>(W3, ws);
  k_cand_t0<<<(C_NUM * D_DIM) / 256, 256, 0, stream>>>(W1, W4, ws);
  k_trans<<<C_NUM, 256, 0, stream>>>(ws);
  k_seq<<<B_SZ, 256, 0, stream>>>(E, W1, pos, db, mh, mk, mb, ws, out);
}
